// CrossAtt_57775900066243
// MI455X (gfx1250) — matrix operand order repaired and checked by reading, not compiled, not run on hardware
//
#include <hip/hip_runtime.h>

typedef unsigned short u16;
typedef unsigned int   u32;
typedef __attribute__((ext_vector_type(16))) __bf16 v16bf;
typedef __attribute__((ext_vector_type(8)))  float  v8f;
typedef __attribute__((ext_vector_type(4)))  int    i4;
typedef __attribute__((ext_vector_type(4)))  unsigned int u32x4;
typedef __attribute__((ext_vector_type(8)))  int    i32x8;
typedef __attribute__((ext_vector_type(4)))  int    i32x4;

union Frag { v16bf v; i4 q[2]; };

__device__ __forceinline__ u16 f2bf(float f) {
  u32 u = __float_as_uint(f);
  u32 r = (u + 0x7FFFu + ((u >> 16) & 1u)) >> 16;  // round-to-nearest-even
  return (u16)r;
}

__device__ __forceinline__ void async_cp16(u32 lds, const void* g) {
  // CDNA5 async global->LDS copy (16B per active lane), tracked by ASYNCcnt
  asm volatile("global_load_async_to_lds_b128 %0, %1, off" :: "v"(lds), "v"(g) : "memory");
}
__device__ __forceinline__ void wait_async0() {
  asm volatile("s_wait_asynccnt 0" ::: "memory");
}

__device__ __forceinline__ v8f wmma_bf16(const Frag& a, const Frag& b, v8f c) {
  return __builtin_amdgcn_wmma_f32_16x16x32_bf16(false, a.v, false, b.v, (short)0, c, false, false);
}

// TDM: DMA a 128x32-element bf16 tile (row stride K elements) from global into
// contiguous 8KB of LDS. D# built per CDNA5 ISA ch.8 (group0 + group1).
__device__ __forceinline__ void tdm_load_A(const u16* g, u32 lds, int K) {
  unsigned long long ga = (unsigned long long)(size_t)g;
  u32x4 g0;
  g0[0] = 1u;                               // count=1 (valid user descriptor)
  g0[1] = lds;                              // lds_addr (bytes)
  g0[2] = (u32)ga;                          // global_addr[31:0]
  g0[3] = (u32)(ga >> 32) | 0x80000000u;    // global_addr[56:32] | type=2<<30
  i32x8 g1;
  g1[0] = 0x00010000;                       // workgroup_mask=0, data_size=1 (2B)
  g1[1] = (int)(32u << 16);                 // tensor_dim0 = 32  (bits 63:48)
  g1[2] = (int)(128u << 16);                // tensor_dim1 = 128 (bits 95:80)
  g1[3] = (int)(32u << 16);                 // tile_dim0  = 32   (bits 127:112)
  g1[4] = 128;                              // tile_dim1  = 128  (bits 143:128)
  g1[5] = K;                                // tensor_dim0_stride[31:0] (elements)
  g1[6] = 0;
  g1[7] = 0;
  i32x4 g2 = {0, 0, 0, 0};
  i32x4 g3 = {0, 0, 0, 0};
#if defined(__clang_major__) && __clang_major__ >= 23
  i32x8 g4 = {0, 0, 0, 0, 0, 0, 0, 0};
  __builtin_amdgcn_tensor_load_to_lds(g0, g1, g2, g3, g4, 0);
#else
  __builtin_amdgcn_tensor_load_to_lds(g0, g1, g2, g3, 0);
#endif
}

// ---------------- prep / conversion kernels ----------------

__global__ void k_cvt_wsa(const float* __restrict__ w, u16* __restrict__ d) {
  int i = blockIdx.x * 256 + threadIdx.x;
  if (i < 512 * 2048) d[i] = f2bf(w[i]);          // [co][ci] row-major already
}

__global__ void k_cvt_wconv(const float* __restrict__ w, u16* __restrict__ d) {
  int i = blockIdx.x * 256 + threadIdx.x;
  if (i >= 512 * 1024 * 9) return;
  int t  = i % 9;                                  // kh*3+kw
  int rc = i / 9;
  int ci = rc & 1023;
  int co = rc >> 10;
  d[(size_t)co * 9216 + t * 1024 + ci] = f2bf(w[i]);  // K-order = (tap, ci)
}

__global__ void k_cvt_visT(const float* __restrict__ vis, u16* __restrict__ d) {
  int i = blockIdx.x * 256 + threadIdx.x;
  if (i >= 2 * 2048 * 256) return;
  int p  = i & 255;
  int ci = (i >> 8) & 2047;
  int b  = i >> 19;
  d[((size_t)(b * 256 + p)) * 2048 + ci] = f2bf(vis[i]);  // [b][p][ci]
}

__global__ void k_lang(const float* __restrict__ emb, const float* __restrict__ wl,
                       const float* __restrict__ bl, u16* __restrict__ d) {
  int i = blockIdx.x * 256 + threadIdx.x;
  if (i >= 2 * 8 * 512) return;
  int e = i & 511;
  int r = i >> 9;                                  // b*8+l
  float acc = bl[e];
  const float* er = emb + (size_t)r * 1000;
  for (int c = 0; c < 1000; ++c) acc += er[c] * wl[c * 512 + e];
  d[i] = f2bf(acc);                                // [b*8+l][e]
}

__global__ void k_multi(const u16* __restrict__ visatt, const u16* __restrict__ langA,
                        u16* __restrict__ d) {
  int i = blockIdx.x * 256 + threadIdx.x;
  if (i >= 16 * 256 * 1024) return;
  int ci  = i & 1023;
  int p   = (i >> 10) & 255;
  int img = i >> 18;
  int b = img >> 3, l = img & 7;
  d[i] = (ci < 512) ? visatt[((size_t)(b * 256 + p)) * 512 + ci]
                    : langA[((size_t)(b * 8 + l)) * 512 + (ci - 512)];
}

// ---------------- generic implicit-GEMM conv (WMMA bf16) ----------------
// C[M=512 co][N=256 px per image] = A[co][K] * B_im2col[px][K], K = taps<<cinShift.
// A tile staged by the Tensor Data Mover, B tile by async global->LDS copies
// with zero-fill for conv padding; LDS double-buffered so DMA hides under WMMA.
// modes: 0 -> bf16 [bh][s][d] (q/k), 2 -> bf16 [bh][d][s] (v), 3 -> f32 [img][co][p] (r),
//        4 -> bf16 [b][p][co] (vis_att, taps==1)
__device__ __forceinline__ void stage_B(const u16* __restrict__ src, size_t srcbase,
                                        u16* Bbuf, int tid, int n0, int k0,
                                        int cinShift, int Cin, int taps) {
  int tap = k0 >> cinShift;
  int ci0 = k0 & (Cin - 1);
  int dy = 0, dx = 0;
  if (taps == 9) { int kh = tap / 3; dy = kh - 1; dx = (tap - kh * 3) - 1; }
#pragma unroll
  for (int c = 0; c < 2; ++c) {
    int idx = tid + c * 256;                 // 0..511 chunks of 16B
    int n = idx >> 2, ko = (idx & 3) << 3;
    int p = n0 + n;
    int px = (p & 15) + dx, py = (p >> 4) + dy;
    u16* dst = &Bbuf[n * 32 + ko];
    if ((u32)px < 16u && (u32)py < 16u) {
      const u16* gp = src + srcbase + (((size_t)(py * 16 + px)) << cinShift) + ci0 + ko;
      async_cp16((u32)(size_t)dst, gp);
    } else {
      i4 z = {0, 0, 0, 0};
      *(i4*)dst = z;                          // conv zero padding
    }
  }
}

__global__ __launch_bounds__(256) void k_gemm(
    const u16* __restrict__ A, const float* __restrict__ bias,
    const u16* __restrict__ src, int cinShift, int taps, int mode,
    u16* __restrict__ outb, float* __restrict__ outf)
{
  const int Cin = 1 << cinShift;
  const int K = taps << cinShift;
  const int iters = K >> 5;
  __shared__ __align__(16) u16 Als[2][128 * 32];
  __shared__ __align__(16) u16 Bls[2][128 * 32];

  const int tid  = threadIdx.x;
  const int lane = tid & 31;
  const int wv   = tid >> 5;
  const int img  = blockIdx.z;
  const int co0  = blockIdx.y * 128;
  const int n0   = blockIdx.x * 128;
  const int wr = wv >> 2, wc = wv & 3;
  const int cl = lane & 15;
  const int hi = (lane >= 16);

  v8f zero8 = {0.f,0.f,0.f,0.f,0.f,0.f,0.f,0.f};
  v8f acc[4][2];
#pragma unroll
  for (int mi = 0; mi < 4; ++mi) { acc[mi][0] = zero8; acc[mi][1] = zero8; }

  const u16* Abase = A + (size_t)co0 * K;
  const size_t srcbase = ((size_t)img * 256) << cinShift;
  const bool issuer = (__builtin_amdgcn_readfirstlane(wv) == 0);  // scalar branch

  // prologue: stage tile 0 into buffer 0
  if (issuer) tdm_load_A(Abase, (u32)(size_t)&Als[0][0], K);
  stage_B(src, srcbase, &Bls[0][0], tid, n0, 0, cinShift, Cin, taps);
  wait_async0();
  __builtin_amdgcn_s_wait_tensorcnt(0);
  __syncthreads();

  for (int it = 0; it < iters; ++it) {
    const int buf = it & 1;
    // prefetch next K-tile into the other buffer while we compute this one
    if (it + 1 < iters) {
      if (issuer) tdm_load_A(Abase + ((it + 1) << 5), (u32)(size_t)&Als[buf ^ 1][0], K);
      stage_B(src, srcbase, &Bls[buf ^ 1][0], tid, n0, (it + 1) << 5, cinShift, Cin, taps);
    }
    // ---- compute 8 WMMA tiles per wave on the current buffer ----
    const u16* Ab = &Als[buf][0];
    const u16* Bb = &Bls[buf][0];
    Frag bf0, bf1;
    {
      int n  = wc * 32 + cl;
      int ks = hi ? 8 : 0;
      bf0.q[0] = *(const i4*)&Bb[n * 32 + ks];
      bf0.q[1] = *(const i4*)&Bb[n * 32 + ks + 16];
      bf1.q[0] = *(const i4*)&Bb[(n + 16) * 32 + ks];
      bf1.q[1] = *(const i4*)&Bb[(n + 16) * 32 + ks + 16];
    }
#pragma unroll
    for (int mi = 0; mi < 4; ++mi) {
      Frag af;
      int row = wr * 64 + mi * 16 + cl;
      int kb = hi ? 8 : 0;
      af.q[0] = *(const i4*)&Ab[row * 32 + kb];
      af.q[1] = *(const i4*)&Ab[row * 32 + kb + 16];
      acc[mi][0] = wmma_bf16(af, bf0, acc[mi][0]);
      acc[mi][1] = wmma_bf16(af, bf1, acc[mi][1]);
    }
    wait_async0();                         // next B tile landed (this wave's copies)
    __builtin_amdgcn_s_wait_tensorcnt(0);  // next A tile landed (issuer; NOP otherwise)
    __syncthreads();
  }

  // ---- store with bias, layout per mode ----
  const int rhi = hi ? 8 : 0;
#pragma unroll
  for (int mi = 0; mi < 4; ++mi)
#pragma unroll
    for (int ni = 0; ni < 2; ++ni)
#pragma unroll
      for (int i = 0; i < 8; ++i) {
        int co = co0 + wr * 64 + mi * 16 + i + rhi;
        int p  = n0 + wc * 32 + ni * 16 + cl;
        float vvv = acc[mi][ni][i] + bias[co];
        if (mode == 0) {                       // q/k: [b*8+h][l*256+p][d]
          int b = img >> 3, l = img & 7;
          int h = co >> 6, d = co & 63;
          outb[(((size_t)(b * 8 + h)) * 2048 + (size_t)(l * 256 + p)) * 64 + d] = f2bf(vvv);
        } else if (mode == 2) {                // v: [b*8+h][d][l*256+p]
          int b = img >> 3, l = img & 7;
          int h = co >> 6, d = co & 63;
          outb[(((size_t)(b * 8 + h)) * 64 + d) * 2048 + (l * 256 + p)] = f2bf(vvv);
        } else if (mode == 3) {                // r: f32 [img][co][p]
          outf[((size_t)img * 512 + co) * 256 + p] = vvv;
        } else {                               // vis_att: [b][p][co]
          outb[((size_t)img * 256 + p) * 512 + co] = f2bf(vvv);
        }
      }
}

// ---------------- flash attention (no S x S materialization) ----------------
// per wave: 16 queries, full d=64; loop KV in chunks of 64 with online softmax.
__global__ __launch_bounds__(256) void k_attn(
    const u16* __restrict__ qb, const u16* __restrict__ kbuf,
    const u16* __restrict__ vb, float* __restrict__ vatt)
{
  __shared__ __align__(16) u16 Pls[8 * 16 * 64];
  const int tid = threadIdx.x, lane = tid & 31, wv = tid >> 5;
  const int bh = blockIdx.y;
  const int t0 = blockIdx.x * 128 + wv * 16;
  const int cl = lane & 15;
  const int hi = (lane >= 16);
  const u32 pbase = wv * 16 * 64;

  v8f zero8 = {0.f,0.f,0.f,0.f,0.f,0.f,0.f,0.f};
  Frag aQ[2];
  {
    const u16* qr = qb + ((size_t)bh * 2048 + t0 + cl) * 64;
    int kbv = hi ? 8 : 0;
#pragma unroll
    for (int ks = 0; ks < 2; ++ks) {
      aQ[ks].q[0] = *(const i4*)&qr[ks * 32 + kbv];
      aQ[ks].q[1] = *(const i4*)&qr[ks * 32 + kbv + 16];
    }
  }
  v8f O[4];
  float m_i[8], l_i[8];
#pragma unroll
  for (int j = 0; j < 4; ++j) O[j] = zero8;
#pragma unroll
  for (int i = 0; i < 8; ++i) { m_i[i] = -3.0e38f; l_i[i] = 0.f; }

  for (int s0 = 0; s0 < 2048; s0 += 64) {
    // scores 16x64 = QK^T
    v8f sc[4];
#pragma unroll
    for (int j = 0; j < 4; ++j) {
      sc[j] = zero8;
      const u16* kr = kbuf + ((size_t)bh * 2048 + s0 + j * 16 + cl) * 64;
      int ks0 = hi ? 8 : 0;
#pragma unroll
      for (int ks = 0; ks < 2; ++ks) {
        Frag bfK;
        bfK.q[0] = *(const i4*)&kr[ks * 32 + ks0];
        bfK.q[1] = *(const i4*)&kr[ks * 32 + ks0 + 16];
        sc[j] = wmma_bf16(aQ[ks], bfK, sc[j]);
      }
    }
    // online softmax (row stats across 16 lanes of each half-wave)
#pragma unroll
    for (int i = 0; i < 8; ++i) {
      float mx = fmaxf(fmaxf(sc[0][i], sc[1][i]), fmaxf(sc[2][i], sc[3][i]));
#pragma unroll
      for (int msk = 1; msk < 16; msk <<= 1) mx = fmaxf(mx, __shfl_xor(mx, msk, 32));
      float mn = fmaxf(m_i[i], mx);
      float al = __expf(m_i[i] - mn);
      float rs = 0.f;
#pragma unroll
      for (int j = 0; j < 4; ++j) { float pv = __expf(sc[j][i] - mn); sc[j][i] = pv; rs += pv; }
#pragma unroll
      for (int msk = 1; msk < 16; msk <<= 1) rs += __shfl_xor(rs, msk, 32);
      l_i[i] = l_i[i] * al + rs;
      m_i[i] = mn;
#pragma unroll
      for (int jd = 0; jd < 4; ++jd) O[jd][i] = O[jd][i] * al;
    }
    // re-layout P: C-frag -> LDS [t][s] -> A-frag
#pragma unroll
    for (int j = 0; j < 4; ++j)
#pragma unroll
      for (int i = 0; i < 8; ++i)
        Pls[pbase + (i + (hi ? 8 : 0)) * 64 + j * 16 + cl] = f2bf(sc[j][i]);
    __syncthreads();
    Frag pA[2];
    {
      const u16* pr = &Pls[pbase + cl * 64];
      int kbv = hi ? 8 : 0;
#pragma unroll
      for (int ks = 0; ks < 2; ++ks) {
        pA[ks].q[0] = *(const i4*)&pr[ks * 32 + kbv];
        pA[ks].q[1] = *(const i4*)&pr[ks * 32 + kbv + 16];
      }
    }
    // O += P * V
#pragma unroll
    for (int jd = 0; jd < 4; ++jd) {
      const u16* vr = vb + ((size_t)bh * 64 + jd * 16 + cl) * 2048 + s0;
      int ks0 = hi ? 8 : 0;
#pragma unroll
      for (int ks = 0; ks < 2; ++ks) {
        Frag bfV;
        bfV.q[0] = *(const i4*)&vr[ks * 32 + ks0];
        bfV.q[1] = *(const i4*)&vr[ks * 32 + ks0 + 16];
        O[jd] = wmma_bf16(pA[ks], bfV, O[jd]);
      }
    }
    __syncthreads();
  }
  // normalize and store v_att as f32 [bh][t][d]
#pragma unroll
  for (int jd = 0; jd < 4; ++jd)
#pragma unroll
    for (int i = 0; i < 8; ++i) {
      int t = t0 + i + (hi ? 8 : 0);
      int d = jd * 16 + cl;
      vatt[((size_t)bh * 2048 + t) * 64 + d] = O[jd][i] / l_i[i];
    }
}

// ---------------- final: out[b][e][p] = mean_l (v_att + r) ----------------
__global__ void k_final(const float* __restrict__ vatt, const float* __restrict__ rbuf,
                        float* __restrict__ out) {
  int i = blockIdx.x * 256 + threadIdx.x;
  if (i >= 2 * 512 * 256) return;
  int p = i & 255;
  int e = (i >> 8) & 511;
  int b = i >> 17;
  int h = e >> 6, d = e & 63;
  float acc = 0.f;
#pragma unroll
  for (int l = 0; l < 8; ++l) {
    acc += vatt[(((size_t)(b * 8 + h)) * 2048 + (l * 256 + p)) * 64 + d];
    acc += rbuf[(((size_t)(b * 8 + l)) * 512 + e) * 256 + p];
  }
  out[i] = acc * 0.125f;
}

// ---------------- launcher ----------------
extern "C" void kernel_launch(void* const* d_in, const int* in_sizes, int n_in,
                              void* d_out, int out_size, void* d_ws, size_t ws_size,
                              hipStream_t stream) {
  const float* vis   = (const float*)d_in[0];
  const float* emb   = (const float*)d_in[1];
  const float* w_sa  = (const float*)d_in[2];
  const float* b_sa  = (const float*)d_in[3];
  const float* w_lin = (const float*)d_in[4];
  const float* b_lin = (const float*)d_in[5];
  const float* w_q   = (const float*)d_in[6];
  const float* b_q   = (const float*)d_in[7];
  const float* w_k   = (const float*)d_in[8];
  const float* b_k   = (const float*)d_in[9];
  const float* w_v   = (const float*)d_in[10];
  const float* b_v   = (const float*)d_in[11];
  const float* w_r   = (const float*)d_in[12];
  const float* b_r   = (const float*)d_in[13];
  float* out = (float*)d_out;

  char* p = (char*)d_ws;
  auto carve = [&](size_t bytes) -> void* {
    void* r = (void*)p;
    p += (bytes + 255) & ~(size_t)255;
    return r;
  };
  u16* wsa_bf = (u16*)carve((size_t)512 * 2048 * 2);
  u16* wq_bf  = (u16*)carve((size_t)512 * 9216 * 2);
  u16* wk_bf  = (u16*)carve((size_t)512 * 9216 * 2);
  u16* wv_bf  = (u16*)carve((size_t)512 * 9216 * 2);
  u16* wr_bf  = (u16*)carve((size_t)512 * 9216 * 2);
  u16* visT   = (u16*)carve((size_t)2 * 256 * 2048 * 2);
  u16* visatt = (u16*)carve((size_t)2 * 256 * 512 * 2);
  u16* langA  = (u16*)carve((size_t)2 * 8 * 512 * 2);
  u16* multi  = (u16*)carve((size_t)16 * 256 * 1024 * 2);
  u16* qbuf   = (u16*)carve((size_t)16 * 2048 * 64 * 2);
  u16* kbuf   = (u16*)carve((size_t)16 * 2048 * 64 * 2);
  u16* vbuf   = (u16*)carve((size_t)16 * 2048 * 64 * 2);
  float* rbuf = (float*)carve((size_t)16 * 512 * 256 * 4);
  float* vatt = (float*)carve((size_t)16 * 2048 * 64 * 4);

  k_cvt_wsa  <<<4096, 256, 0, stream>>>(w_sa, wsa_bf);
  k_cvt_wconv<<<18432, 256, 0, stream>>>(w_q, wq_bf);
  k_cvt_wconv<<<18432, 256, 0, stream>>>(w_k, wk_bf);
  k_cvt_wconv<<<18432, 256, 0, stream>>>(w_v, wv_bf);
  k_cvt_wconv<<<18432, 256, 0, stream>>>(w_r, wr_bf);
  k_cvt_visT <<<4096, 256, 0, stream>>>(vis, visT);
  k_lang     <<<32, 256, 0, stream>>>(emb, w_lin, b_lin, langA);

  // vis_att = 1x1 conv: M=512, N=256 per batch, K=2048
  k_gemm<<<dim3(2, 4, 2), 256, 0, stream>>>(wsa_bf, b_sa, visT, 11, 1, 4, visatt, nullptr);
  k_multi<<<16384, 256, 0, stream>>>(visatt, langA, multi);

  // 3x3 convs: M=512, N=256 per image, K=9216
  k_gemm<<<dim3(2, 4, 16), 256, 0, stream>>>(wq_bf, b_q, multi, 10, 9, 0, qbuf, nullptr);
  k_gemm<<<dim3(2, 4, 16), 256, 0, stream>>>(wk_bf, b_k, multi, 10, 9, 0, kbuf, nullptr);
  k_gemm<<<dim3(2, 4, 16), 256, 0, stream>>>(wv_bf, b_v, multi, 10, 9, 2, vbuf, nullptr);
  k_gemm<<<dim3(2, 4, 16), 256, 0, stream>>>(wr_bf, b_r, multi, 10, 9, 3, nullptr, rbuf);

  k_attn <<<dim3(16, 16), 256, 0, stream>>>(qbuf, kbuf, vbuf, vatt);
  k_final<<<1024, 256, 0, stream>>>(vatt, rbuf, out);
  (void)in_sizes; (void)n_in; (void)out_size; (void)ws_size;
}
